// LSTMsep_37606733643972
// MI455X (gfx1250) — matrix operand order repaired and checked by reading, not compiled, not run on hardware
//
#include <hip/hip_runtime.h>
#include <hip/hip_bf16.h>
#include <math.h>

// ---------------------------------------------------------------------------
// Multi-head LSTM (8 heads, hid=64) + per-head Linear + cross-head sum.
// grid = HEADS(8) x BATCH_CHUNKS(4) persistent WGs, block = 128 (4 waves).
// Wave w owns hid slice [16w,16w+16): it computes the i/f/g/o gate tiles for
// its slice with 12 v_wmma_f32_16x16x32_f16 (weights live in VGPRs), so the
// gate nonlinearity runs straight out of the accumulators.
// Two split LDS-only barriers per step, both waits hidden behind compute:
//   signal#1 after A-frag loads  -> wait after WMMA+gates (protects A reuse)
//   signal#2 after h/x stores    -> wait after atomic/reduce tail (A ready)
// Activations use hardware v_tanh_f32 (TRANS).
// ---------------------------------------------------------------------------

typedef __attribute__((ext_vector_type(16))) _Float16 v16h;
typedef __attribute__((ext_vector_type(8)))  float    v8f;

#define T_SEQ   2048
#define NBATCH  64
#define NIN     16
#define NHEADS  8
#define NHID    64
#define NGATES  256          // 4*NHID
#define BCHUNK  16           // batch rows per workgroup (one WMMA M tile)
#define NBC     (NBATCH / BCHUNK)
#define LDK     98           // K row stride in halves (49 dwords -> conflict-free)

#define AB_BYTES   (BCHUNK * LDK * 2)          // 3136 B: f16 A [16][LDK]
#define SMEM_BYTES (AB_BYTES + 64)

// Split LDS-only barrier halves: never wait on vmem (keeps prefetch in flight).
__device__ __forceinline__ void barrier_signal_lds() {
  asm volatile("s_wait_dscnt 0x0\n\ts_barrier_signal -1" ::: "memory");
}
__device__ __forceinline__ void barrier_wait_wg() {
  asm volatile("s_barrier_wait -1" ::: "memory");
}

// Hardware tanh (CDNA5 TRANS op). Builtin if declared, else raw ISA with a
// v_nop to satisfy the TRANS result-use hazard (ISA 7.4) by hand.
__device__ __forceinline__ float hw_tanh(float x) {
#if __has_builtin(__builtin_amdgcn_tanhf)
  return __builtin_amdgcn_tanhf(x);
#else
  float r;
  asm("v_tanh_f32 %0, %1\n\tv_nop" : "=v"(r) : "v"(x));
  return r;
#endif
}
__device__ __forceinline__ float hw_sigmoid(float x) {
  return fmaf(0.5f, hw_tanh(0.5f * x), 0.5f);   // sigma(x) = (1+tanh(x/2))/2
}

// --- A fragment (ISA 7.12.2 16-bit A 16x32 layout) -------------------------
__device__ __forceinline__ v16h load_fragA(const _Float16* base, int k0, int half, int l16) {
  const _Float16* rp = base + l16 * LDK;
  v16h f;
#pragma unroll
  for (int v = 0; v < 8; ++v) {
    int k = (v < 4) ? (k0 + half * 8 + 2 * v) : (k0 + 16 + half * 8 + 2 * (v - 4));
    f[2 * v]     = rp[k];
    f[2 * v + 1] = rp[k + 1];
  }
  return f;
}

// --- B fragment built in registers straight from global f32 weights --------
// Wcat[n][k] = W_ih[n][k] (k<16) | W_hh[n][k-16] (k<80) | 0.
__device__ __forceinline__ v16h build_fragB(const float* __restrict__ wih,
                                            const float* __restrict__ whh,
                                            int n0, int k0, int half, int l16) {
  const int n = n0 + l16;
  v16h f;
#pragma unroll
  for (int v = 0; v < 8; ++v) {
#pragma unroll
    for (int e = 0; e < 2; ++e) {
      int kk = k0 + ((v < 4) ? (half * 8 + 2 * v) : (16 + half * 8 + 2 * (v - 4))) + e;
      float val = (kk < NIN) ? wih[n * NIN + kk]
                : (kk < NIN + NHID) ? whh[n * NHID + (kk - NIN)]
                : 0.0f;
      f[2 * v + e] = (_Float16)val;
    }
  }
  return f;
}

// ---------------------------------------------------------------------------
// prologue: out <- b_lin[h] broadcast, lstm_output <- 0 (both atomically
// accumulated by the main kernel; re-initialized every launch)
__global__ void lstm_init_kernel(float* p, const float* b_lin, int n_out, int n_total) {
  int i = blockIdx.x * blockDim.x + threadIdx.x;
  if (i < n_total) p[i] = (i < n_out) ? b_lin[i & (NHEADS - 1)] : 0.0f;
}

__global__ __launch_bounds__(128)
void lstm_head_kernel(const float* __restrict__ x,
                      const float* __restrict__ W_ih,
                      const float* __restrict__ W_hh,
                      const float* __restrict__ b_ih,
                      const float* __restrict__ b_hh,
                      const float* __restrict__ W_lin,
                      const float* __restrict__ b_lin,
                      float* __restrict__ out,        // (T,B,H)  pre-filled b_lin
                      float* __restrict__ lstm_out)   // (T,B,NHID) pre-zeroed
{
  extern __shared__ char smem[];
  _Float16* Ab = (_Float16*)smem;         // [BCHUNK][LDK] f16 A matrix

  const int tid  = threadIdx.x;
  const int head = blockIdx.x >> 2;       // 0..7
  const int bc   = blockIdx.x & 3;        // 0..3
  const int b0g  = bc * BCHUNK;

  const int lane = tid & 31;
  const int wave = tid >> 5;              // 0..3
  const int half = lane >> 4;
  const int l16  = lane & 15;
  const int k0   = wave * 16;             // this wave's hid slice base
  const int kk   = k0 + l16;              // this lane's hid index

  const float* wih = W_ih + (size_t)head * NGATES * NIN;
  const float* whh = W_hh + (size_t)head * NGATES * NHID;

  // ---- weights -> registers (12 B fragments: 4 gates x 3 K-slabs) --------
  v16h B[4][3];
#pragma unroll
  for (int g = 0; g < 4; ++g)
#pragma unroll
    for (int s = 0; s < 3; ++s)
      B[g][s] = build_fragB(wih, whh, g * NHID + k0, s * 32, half, l16);

  // per-lane constants
  const float bi = b_ih[head * NGATES + kk]       + b_hh[head * NGATES + kk];
  const float bf = b_ih[head * NGATES + 64 + kk]  + b_hh[head * NGATES + 64 + kk];
  const float bg = b_ih[head * NGATES + 128 + kk] + b_hh[head * NGATES + 128 + kk];
  const float bo = b_ih[head * NGATES + 192 + kk] + b_hh[head * NGATES + 192 + kk];
  const float wl = W_lin[head * NHID + kk];

  // ---- zero h (+pad) region of A: cols 16..LDK-1 -> h0 = 0 ---------------
  for (int idx = tid; idx < BCHUNK * (LDK - NIN); idx += 128) {
    int r = idx / (LDK - NIN), k = NIN + idx % (LDK - NIN);
    Ab[r * LDK + k] = (_Float16)0.0f;
  }

  // ---- x staging: thread covers rows r0 and r0+8, col i0 -----------------
  const int r0 = tid >> 4;                // 0..7
  const int i0 = tid & 15;
  const float* xp = x + ((size_t)b0g + r0) * NIN + i0;
  // stage x(0) now; keep x(1) prefetched in regs
  Ab[r0 * LDK + i0]       = (_Float16)xp[0];
  Ab[(r0 + 8) * LDK + i0] = (_Float16)xp[8 * NIN];
  xp += (size_t)NBATCH * NIN;
  float xreg0 = xp[0];
  float xreg1 = xp[8 * NIN];

  // cell state: lane owns (b = v + 8*half, k = kk), v = 0..7
  float creg[8] = {0.f, 0.f, 0.f, 0.f, 0.f, 0.f, 0.f, 0.f};

  __syncthreads();                        // A(0) ready

  for (int t = 0; t < T_SEQ; ++t) {
    // -- A(t) fragments ----------------------------------------------------
    v16h a0 = load_fragA(Ab, 0,  half, l16);
    v16h a1 = load_fragA(Ab, 32, half, l16);
    v16h a2 = load_fragA(Ab, 64, half, l16);
    barrier_signal_lds();                 // my A reads complete

    // -- z-slice = A * Wcat^T for this wave's 4 gate tiles -----------------
    v8f acc[4];
#pragma unroll
    for (int g = 0; g < 4; ++g) {
      v8f z = {};
      z = __builtin_amdgcn_wmma_f32_16x16x32_f16(false, a0, false, B[g][0], (short)0, z, false, false);
      z = __builtin_amdgcn_wmma_f32_16x16x32_f16(false, a1, false, B[g][1], (short)0, z, false, false);
      z = __builtin_amdgcn_wmma_f32_16x16x32_f16(false, a2, false, B[g][2], (short)0, z, false, false);
      acc[g] = z;
    }

    // -- gates straight from accumulators (PyTorch order i,f,g,o) ---------
    float hv[8], red[8];
#pragma unroll
    for (int v = 0; v < 8; ++v) {
      float ig = hw_sigmoid(acc[0][v] + bi);
      float fg = hw_sigmoid(acc[1][v] + bf);
      float gg = hw_tanh   (acc[2][v] + bg);
      float og = hw_sigmoid(acc[3][v] + bo);
      float c  = fmaf(fg, creg[v], ig * gg);
      creg[v]  = c;
      hv[v]    = og * hw_tanh(c);
      red[v]   = hv[v] * wl;
    }

    barrier_wait_wg();                    // all waves done reading A(t)

    // -- write h(t+1) + stage x(t+1), prefetch x(t+2) ----------------------
#pragma unroll
    for (int v = 0; v < 8; ++v)
      Ab[(v + 8 * half) * LDK + NIN + kk] = (_Float16)hv[v];
    Ab[r0 * LDK + i0]       = (_Float16)xreg0;
    Ab[(r0 + 8) * LDK + i0] = (_Float16)xreg1;
    {
      const size_t adv = (t + 2 < T_SEQ) ? (size_t)NBATCH * NIN : 0;  // no peel
      xp += adv;
      xreg0 = xp[0];
      xreg1 = xp[8 * NIN];
    }
    barrier_signal_lds();                 // my A(t+1) writes visible

    // -- tail hidden behind the barrier: atomics + Linear reduction --------
#pragma unroll
    for (int v = 0; v < 8; ++v)
      atomicAdd(&lstm_out[((size_t)t * NBATCH + b0g + v + 8 * half) * NHID + kk], hv[v]);
#pragma unroll
    for (int m = 8; m >= 1; m >>= 1)
#pragma unroll
      for (int v = 0; v < 8; ++v)
        red[v] += __shfl_xor(red[v], m, 16);
    if (l16 == 0) {
#pragma unroll
      for (int v = 0; v < 8; ++v)
        atomicAdd(&out[((size_t)t * NBATCH + b0g + v + 8 * half) * NHEADS + head], red[v]);
    }

    barrier_wait_wg();                    // A(t+1) ready for next iteration
  }
}

// ---------------------------------------------------------------------------
extern "C" void kernel_launch(void* const* d_in, const int* in_sizes, int n_in,
                              void* d_out, int out_size, void* d_ws, size_t ws_size,
                              hipStream_t stream) {
  const float* x     = (const float*)d_in[0];
  const float* W_ih  = (const float*)d_in[1];
  const float* W_hh  = (const float*)d_in[2];
  const float* b_ih  = (const float*)d_in[3];
  const float* b_hh  = (const float*)d_in[4];
  const float* W_lin = (const float*)d_in[5];
  const float* b_lin = (const float*)d_in[6];

  float* out = (float*)d_out;                                 // (T,B,H) then (T,B,NHID)
  float* lstm = out + (size_t)T_SEQ * NBATCH * NHEADS;

  // both output regions are atomically accumulated -> re-init every launch
  const int n_out   = T_SEQ * NBATCH * NHEADS;
  const int n_total = n_out + T_SEQ * NBATCH * NHID;
  lstm_init_kernel<<<dim3((n_total + 255) / 256), dim3(256), 0, stream>>>(
      out, b_lin, n_out, n_total);

  lstm_head_kernel<<<dim3(NHEADS * NBC), dim3(128), SMEM_BYTES, stream>>>(
      x, W_ih, W_hh, b_ih, b_hh, W_lin, b_lin, out, lstm);
}
